// SHGD_43241730736177
// MI455X (gfx1250) — compile-verified, matrix operand order checked by reading, not run on hardware
//
#include <hip/hip_runtime.h>
#include <hip/hip_bf16.h>

// ---------------- problem constants (from reference) ----------------
#define BB     256       // batch
#define II     10000     // items
#define UU     20000     // users
#define EE     200000    // entities
#define DD     64        // embedding dim
#define NNZ    400000
#define KG     1200000
// ALPHA / T_MAX = 2.5 / 2.0
#define ALPHA_OVER_T 1.25f

typedef __attribute__((ext_vector_type(16))) __bf16 v16bf;
typedef __attribute__((ext_vector_type(8)))  float  v8f;

#if __has_builtin(__builtin_amdgcn_tanhf)
#define FAST_TANH(x) __builtin_amdgcn_tanhf(x)   // native V_TANH_F32 on gfx1250
#else
#define FAST_TANH(x) tanhf(x)
#endif

__device__ __forceinline__ v8f wmma_bf16(v16bf a, v16bf b, v8f c) {
    // D = A(16x32 bf16) x B(32x16 bf16) + C(16x16 f32)
    return __builtin_amdgcn_wmma_f32_16x16x32_bf16(
        /*neg_a=*/false, a, /*neg_b=*/false, b,
        /*c_mod=*/(short)0, c, /*reuse_a=*/false, /*reuse_b=*/false);
}

__device__ __forceinline__ v16bf cvt16(const float* t) {
    v16bf f;
#pragma unroll
    for (int j = 0; j < 16; ++j) f[j] = (__bf16)t[j];
    return f;
}

// Load one bf16 A/B fragment from a contiguous 64-float row at 32-bit offset.
// Fragment elements 0..7  = row[base + 0..7]   (K = kb*32 + is_hi*8 + j)
// Fragment elements 8..15 = row[base+16 + 0..7]
__device__ __forceinline__ v16bf row_frag(const float* __restrict__ p,
                                          unsigned off) {
    float t[16];
    float4 a = *(const float4*)(p + off);
    float4 b = *(const float4*)(p + off + 4);
    float4 c = *(const float4*)(p + off + 16);
    float4 d = *(const float4*)(p + off + 20);
    t[0]=a.x;  t[1]=a.y;  t[2]=a.z;  t[3]=a.w;
    t[4]=b.x;  t[5]=b.y;  t[6]=b.z;  t[7]=b.w;
    t[8]=c.x;  t[9]=c.y;  t[10]=c.z; t[11]=c.w;
    t[12]=d.x; t[13]=d.y; t[14]=d.z; t[15]=d.w;
    return cvt16(t);
}

// Same, but fragment = (dst_row - src_row) in bf16 (32-bit row offsets).
__device__ __forceinline__ v16bf delta_frag(const float* __restrict__ emb,
                                            unsigned doff, unsigned soff,
                                            unsigned base) {
    float t[16];
    float4 d0 = *(const float4*)(emb + doff + base);
    float4 d1 = *(const float4*)(emb + doff + base + 4);
    float4 d2 = *(const float4*)(emb + doff + base + 16);
    float4 d3 = *(const float4*)(emb + doff + base + 20);
    float4 s0 = *(const float4*)(emb + soff + base);
    float4 s1 = *(const float4*)(emb + soff + base + 4);
    float4 s2 = *(const float4*)(emb + soff + base + 16);
    float4 s3 = *(const float4*)(emb + soff + base + 20);
    t[0]=d0.x-s0.x;  t[1]=d0.y-s0.y;  t[2]=d0.z-s0.z;  t[3]=d0.w-s0.w;
    t[4]=d1.x-s1.x;  t[5]=d1.y-s1.y;  t[6]=d1.z-s1.z;  t[7]=d1.w-s1.w;
    t[8]=d2.x-s2.x;  t[9]=d2.y-s2.y;  t[10]=d2.z-s2.z; t[11]=d2.w-s2.w;
    t[12]=d3.x-s3.x; t[13]=d3.y-s3.y; t[14]=d3.z-s3.z; t[15]=d3.w-s3.w;
    return cvt16(t);
}

// ---------------- kernel 1/4: LDS-tiled transpose  src[R][C] -> dst[C][R] ----
__global__ __launch_bounds__(256) void transpose_k(const float* __restrict__ src,
                                                   float* __restrict__ dst,
                                                   int R, int C) {
    __shared__ float tile[32][33];
    int c0 = blockIdx.x * 32, r0 = blockIdx.y * 32;
#pragma unroll
    for (int k = 0; k < 4; ++k) {
        int r = r0 + threadIdx.y + k * 8;
        int c = c0 + threadIdx.x;
        if (r < R && c < C) tile[threadIdx.y + k * 8][threadIdx.x] = src[r * C + c];
    }
    __syncthreads();
#pragma unroll
    for (int k = 0; k < 4; ++k) {
        int c = c0 + threadIdx.y + k * 8;
        int r = r0 + threadIdx.x;
        if (r < R && c < C) dst[c * R + r] = tile[threadIdx.x][threadIdx.y + k * 8];
    }
}

// ---------------- kernels 2/3: COO SpMM scatter (dense dim = 256) -----------
// out[idx_out[e]][tid] += vals[e] * in[idx_in[e]][tid]
__global__ __launch_bounds__(256) void spmm_scatter(const int* __restrict__ idx_out,
                                                    const int* __restrict__ idx_in,
                                                    const float* __restrict__ vals,
                                                    const float* __restrict__ dense_in,
                                                    float* __restrict__ dense_out,
                                                    int nnz, int edges_per_block) {
    unsigned tid = threadIdx.x;            // 0..255, one dense column each
    int e0 = blockIdx.x * edges_per_block;
    for (int k = 0; k < edges_per_block; ++k) {
        int e = e0 + k;
        if (e >= nnz) return;
        unsigned r = (unsigned)idx_out[e];
        unsigned c = (unsigned)idx_in[e];
        float v = vals[e];
        atomicAdd(&dense_out[r * BB + tid], v * dense_in[c * BB + tid]);
    }
}

// ---------------- kernel 5: divergence gate, WMMA bf16 ----------------------
// 16 edges per wave-tile: V = tanh(delta[16x64] @ W_vel[64x64]) ; scatter V*xd
#define DIV_WAVES 8
__global__ __launch_bounds__(DIV_WAVES * 32, 1)
void divergence_wmma(const float* __restrict__ emb,
                     const float* __restrict__ Wv,
                     const int*   __restrict__ src,
                     const int*   __restrict__ dst,
                     float*       __restrict__ second,
                     int ntiles) {
    int lane      = threadIdx.x & 31;
    int wave      = threadIdx.x >> 5;
    unsigned m_lo = lane & 15;
    unsigned is_hi = lane >> 4;
    unsigned kbase = is_hi * 8;

    // Preload W_vel B-fragments (loop-invariant): bfrag[n_tile][k_block]
    // B[k][n] = Wv[k*64 + n], lane holds column n = nt*16 + m_lo.
    v16bf bfrag[4][2];
#pragma unroll
    for (int nt = 0; nt < 4; ++nt) {
#pragma unroll
        for (int kb = 0; kb < 2; ++kb) {
            float t[16];
#pragma unroll
            for (int j = 0; j < 8; ++j) {
                unsigned K = kb * 32 + kbase + j;
                t[j]     = Wv[K * 64 + nt * 16 + m_lo];
                t[j + 8] = Wv[(K + 16) * 64 + nt * 16 + m_lo];
            }
            bfrag[nt][kb] = cvt16(t);
        }
    }

    for (int tile = blockIdx.x * DIV_WAVES + wave; tile < ntiles;
         tile += gridDim.x * DIV_WAVES) {
        int e  = tile * 16 + m_lo;          // this lane's edge (A-row M = m_lo)
        int si = src[e];
        int di = dst[e];

        // broadcast the tile's 16 edge row-offsets once (32-bit, GVS addressing)
        unsigned smo[8], dmo[8];
#pragma unroll
        for (int v = 0; v < 8; ++v) {
            int m = v + (int)is_hi * 8;
            smo[v] = (unsigned)__shfl(si, m, 32) * (unsigned)DD;
            dmo[v] = (unsigned)__shfl(di, m, 32) * (unsigned)DD;
        }

        unsigned soff = (unsigned)si * DD;
        unsigned doff = (unsigned)di * DD;
        v16bf a0 = delta_frag(emb, doff, soff, kbase);        // K = 0..31
        v16bf a1 = delta_frag(emb, doff, soff, 32 + kbase);   // K = 32..63

        // one N-tile at a time: single live accumulator, epilogue fused
#pragma unroll
        for (int nt = 0; nt < 4; ++nt) {
            v8f c = {};
            c = wmma_bf16(a0, bfrag[nt][0], c);
            c = wmma_bf16(a1, bfrag[nt][1], c);
            unsigned n = nt * 16 + m_lo;
#pragma unroll
            for (int v = 0; v < 8; ++v) {
                float g   = FAST_TANH(c[v]);
                float xdv = emb[dmo[v] + n];
                atomicAdd(&second[smo[v] + n], g * xdv);
            }
        }
    }
}

// ---------------- kernel 6: projection GEMM (WMMA) + filter combine ---------
// out[b][i] = x + 1.25*t[b]*(prop[b][i]/eigen0 - x) + secA[b]·W_second[i] + b_second[i]
__global__ __launch_bounds__(256, 1)
void final_wmma(const float* __restrict__ x,
                const int*   __restrict__ bids,
                const int*   __restrict__ tvec,
                const float* __restrict__ second,
                const float* __restrict__ Ws,
                const float* __restrict__ bsec,
                const float* __restrict__ prop,
                const float* __restrict__ eigen0,
                float*       __restrict__ out) {
    int lane       = threadIdx.x & 31;
    int wave       = threadIdx.x >> 5;
    unsigned m_lo  = lane & 15;
    unsigned is_hi = lane >> 4;
    unsigned kbase = is_hi * 8;

    int it = blockIdx.x * 8 + wave;          // i-tile index, 625 total
    if (it >= II / 16) return;               // wave-uniform
    unsigned i0 = it * 16;
    unsigned b0 = blockIdx.y * 16;

    float inv_eig = 1.0f / eigen0[0];

    // A: rows of gathered second matrix (M = batch row)
    unsigned bid = (unsigned)bids[b0 + m_lo];
    v16bf a0 = row_frag(second, bid * DD + kbase);
    v16bf a1 = row_frag(second, bid * DD + 32 + kbase);

    // B: B[k][n] = W_second[i0+n][k] -> lane n reads row (i0+m_lo) contiguously
    unsigned wrow = (i0 + m_lo) * DD;
    v16bf w0 = row_frag(Ws, wrow + kbase);
    v16bf w1 = row_frag(Ws, wrow + 32 + kbase);

    v8f acc = {};
    acc = wmma_bf16(a0, w0, acc);
    acc = wmma_bf16(a1, w1, acc);

    unsigned i = i0 + m_lo;
    float bs = bsec[i];
#pragma unroll
    for (int v = 0; v < 8; ++v) {
        unsigned b  = b0 + v + is_hi * 8;
        float xv = x[b * II + i];
        float Ax = prop[b * II + i] * inv_eig;
        float tf = (float)tvec[b];
        float z  = xv + ALPHA_OVER_T * tf * (Ax - xv);
        out[b * II + i] = z + acc[v] + bs;
    }
}

// ---------------------------- launcher --------------------------------------
extern "C" void kernel_launch(void* const* d_in, const int* in_sizes, int n_in,
                              void* d_out, int out_size, void* d_ws, size_t ws_size,
                              hipStream_t stream) {
    const float* x        = (const float*)d_in[0];   // [256,10000]
    const int*   bids     = (const int*)  d_in[1];   // [256]
    const int*   tvec     = (const int*)  d_in[2];   // [256,1]
    const float* emb      = (const float*)d_in[3];   // [200000,64]
    const float* Wv       = (const float*)d_in[4];   // [64,64]
    const float* Ws       = (const float*)d_in[5];   // [10000,64]
    const float* bsec     = (const float*)d_in[6];   // [10000]
    const int*   adj_rows = (const int*)  d_in[7];   // [400000]
    const int*   adj_cols = (const int*)  d_in[8];
    const float* adj_vals = (const float*)d_in[9];
    const int*   src      = (const int*)  d_in[10];  // [1200000]
    const int*   dst      = (const int*)  d_in[11];
    const float* eigen0   = (const float*)d_in[12];  // [1]
    float*       out      = (float*)d_out;           // [256,10000]

    // workspace layout (bytes):
    //   [0,            10,240,000)  xT [I,B]   -- reused later as prop [B,I]
    //   [10,240,000,   30,720,000)  h  [U,B]
    //   [30,720,000,   40,960,000)  propT [I,B]
    //   [40,960,000,   92,160,000)  second [E,D]
    char*  w      = (char*)d_ws;
    float* xT     = (float*)(w);
    float* prop   = (float*)(w);                         // alias (xT dead by then)
    float* h      = (float*)(w + 10240000);
    float* propT  = (float*)(w + 30720000);
    float* sec    = (float*)(w + 40960000);

    // zero the accumulation buffers (graph-capture safe)
    hipMemsetAsync(h,     0, (size_t)UU * BB * 4, stream);
    hipMemsetAsync(propT, 0, (size_t)II * BB * 4, stream);
    hipMemsetAsync(sec,   0, (size_t)EE * DD * 4, stream);

    // 1) xT = x.T  (so SpMM gathers contiguous 1KB rows)
    {
        dim3 grid((II + 31) / 32, (BB + 31) / 32);
        dim3 block(32, 8);
        transpose_k<<<grid, block, 0, stream>>>(x, xT, BB, II);
    }
    // 2) h[row] += val * xT[col]
    {
        const int EPB = 8;
        spmm_scatter<<<NNZ / EPB, BB, 0, stream>>>(adj_rows, adj_cols, adj_vals,
                                                   xT, h, NNZ, EPB);
    }
    // 3) propT[col] += val * h[row]
    {
        const int EPB = 8;
        spmm_scatter<<<NNZ / EPB, BB, 0, stream>>>(adj_cols, adj_rows, adj_vals,
                                                   h, propT, NNZ, EPB);
    }
    // 4) prop = propT.T  (coalesced reads in the combine kernel)
    {
        dim3 grid((BB + 31) / 32, (II + 31) / 32);
        dim3 block(32, 8);
        transpose_k<<<grid, block, 0, stream>>>(propT, prop, II, BB);
    }
    // 5) divergence gate: 75000 tiles of 16 edges, WMMA bf16
    {
        divergence_wmma<<<600, DIV_WAVES * 32, 0, stream>>>(emb, Wv, src, dst, sec,
                                                            KG / 16);
    }
    // 6) projection GEMM + filter + combine
    {
        dim3 grid((II / 16 + 7) / 8, BB / 16);   // (79, 16)
        final_wmma<<<grid, 256, 0, stream>>>(x, bids, tvec, sec, Ws, bsec,
                                             prop, eigen0, out);
    }
}
